// SAModuleMSG_47579647705387
// MI455X (gfx1250) — matrix operand order repaired and checked by reading, not compiled, not run on hardware
//
#include <hip/hip_runtime.h>
#include <hip/hip_bf16.h>

// ---------------------------------------------------------------- constants
#define Bc     8
#define NPER   4096
#define MPER   1024
#define NCENT  (Bc * MPER)      // 8192
#define KNBR   64
#define COUT_TOTAL 320

typedef __attribute__((ext_vector_type(16))) _Float16 v16h;
typedef __attribute__((ext_vector_type(8)))  _Float16 v8h;
typedef __attribute__((ext_vector_type(8)))  float    v8f;
typedef __attribute__((ext_vector_type(4)))  unsigned int v4u;
typedef __attribute__((ext_vector_type(8)))  int      v8i;
typedef __attribute__((ext_vector_type(4)))  int      v4i;

// per-branch/layer tables (branch = radius index)
constexpr int CinT [3][3] = {{ 6,32,32},{ 6,64,64},{ 6,64,96}};
constexpr int KpadT[3][3] = {{32,32,32},{32,64,64},{32,64,96}};
constexpr int CoutT[3][3] = {{32,32,64},{64,64,128},{64,96,128}};
// packed f16 weight offsets (halves), K padded to KpadT; stored SWIZZLED:
// per (col-tile ct, k-step ks) fragment of 32 lanes x 16 halves, lane-contiguous.
constexpr int WoffT[3][3] = {{0,1024,2048},{4096,6144,10240},{18432,20480,26624}};
constexpr int WPACK_HALVES = 38912;
constexpr int CoffT[3] = {0, 64, 192};
__device__ constexpr float R2T[3] = {0.01f, 0.04f, 0.16f};

// ================================================================ FPS
// one workgroup per batch; 256 threads, 16 points each; wave32 shuffles
__global__ __launch_bounds__(256)
void fps_kernel(const float* __restrict__ pos, float* __restrict__ posOut,
                float* __restrict__ batchOut) {
  __shared__ float sx[NPER], sy[NPER], sz[NPER];
  __shared__ float redD[8];
  __shared__ int   redI[8];

  const int b    = blockIdx.x;
  const int tid  = threadIdx.x;
  const int lane = tid & 31, wave = tid >> 5;
  const float* pb = pos + (size_t)b * NPER * 3;

  for (int i = tid; i < NPER; i += 256) {
    sx[i] = pb[i * 3 + 0]; sy[i] = pb[i * 3 + 1]; sz[i] = pb[i * 3 + 2];
  }
  __syncthreads();

  float px[16], py[16], pz[16], mind[16];
  int   pid[16];
#pragma unroll
  for (int k = 0; k < 16; ++k) {
    int i = tid + k * 256;
    pid[k] = i; px[k] = sx[i]; py[k] = sy[i]; pz[k] = sz[i];
    mind[k] = __builtin_inff();
  }

  if (tid == 0) {
    int c = b * MPER;
    posOut[c * 3 + 0] = sx[0]; posOut[c * 3 + 1] = sy[0]; posOut[c * 3 + 2] = sz[0];
    batchOut[c] = (float)b;
  }

  int last = 0;
  for (int m = 1; m < MPER; ++m) {
    const float lx = sx[last], ly = sy[last], lz = sz[last];
    float bd = -1.0f; int bi = 0x7fffffff;
#pragma unroll
    for (int k = 0; k < 16; ++k) {
      float dx = px[k] - lx, dy = py[k] - ly, dz = pz[k] - lz;
      float d  = dx * dx + dy * dy + dz * dz;
      mind[k]  = fminf(mind[k], d);
      if (mind[k] > bd || (mind[k] == bd && pid[k] < bi)) { bd = mind[k]; bi = pid[k]; }
    }
    // wave32 argmax reduce (first-index tiebreak)
#pragma unroll
    for (int off = 16; off > 0; off >>= 1) {
      float od = __shfl_xor(bd, off, 32);
      int   oi = __shfl_xor(bi, off, 32);
      if (od > bd || (od == bd && oi < bi)) { bd = od; bi = oi; }
    }
    if (lane == 0) { redD[wave] = bd; redI[wave] = bi; }
    __syncthreads();
    float wd = redD[0]; int wi = redI[0];
#pragma unroll
    for (int w = 1; w < 8; ++w)
      if (redD[w] > wd || (redD[w] == wd && redI[w] < wi)) { wd = redD[w]; wi = redI[w]; }
    __syncthreads();   // protect redD/redI reuse next iteration
    last = wi;
    if (tid == 0) {
      int c = b * MPER + m;
      posOut[c * 3 + 0] = sx[wi]; posOut[c * 3 + 1] = sy[wi]; posOut[c * 3 + 2] = sz[wi];
      batchOut[c] = (float)b;
    }
  }
}

// ================================================================ weight prep
// Writes wpack in B-fragment order: element index e decomposes as
//   i = e & 15  (half within lane), l = (e>>4) & 31 (lane), f = e >> 9 (fragment)
//   ct = f / nks, ks = f % nks, n = l & 15, khalf = (l>>4)*16
//   k = ks*32 + khalf + i, col = ct*16 + n
// so each lane's v16h B-fragment is 32 contiguous bytes in LDS.
struct W9 { const float* w[9]; };

__global__ __launch_bounds__(256)
void prep_weights(W9 w9, _Float16* __restrict__ wpack) {
  const int g  = blockIdx.x;           // 0..8
  const int br = g / 3, l = g % 3;
  const int Cin = CinT[br][l], Cn = CoutT[br][l], Kp = KpadT[br][l];
  const int nks = Kp >> 5;
  const float* W = w9.w[g];
  _Float16* dst  = wpack + WoffT[br][l];
  for (int e = threadIdx.x; e < Kp * Cn; e += 256) {
    int i   = e & 15;
    int ln  = (e >> 4) & 31;
    int f   = e >> 9;
    int ct  = f / nks;
    int ks  = f - ct * nks;
    int k   = ks * 32 + ((ln >> 4) * 16) + i;
    int col = ct * 16 + (ln & 15);
    float v = (k < Cin) ? W[k * Cn + col] : 0.0f;
    dst[e] = (_Float16)v;
  }
}

// ================================================================ kNN (K=64)
// one workgroup (128 thr) per centroid; binary-search on float bit patterns
// for the 64th-smallest d2, then atomic compaction of the winner set.
__global__ __launch_bounds__(128)
void knn_kernel(const float* __restrict__ pos, const float* __restrict__ posOut,
                int* __restrict__ nbrIdx, float* __restrict__ nbrD2) {
  __shared__ float d2s[NPER];
  __shared__ int   redC[4];
  __shared__ int   cnt1, cnt2;

  const int c = blockIdx.x;
  const int b = c >> 10;
  const int tid = threadIdx.x, lane = tid & 31, wave = tid >> 5;

  const float cx = posOut[c * 3 + 0], cy = posOut[c * 3 + 1], cz = posOut[c * 3 + 2];
  const float* pb = pos + (size_t)b * NPER * 3;
  for (int i = tid; i < NPER; i += 128) {
    float dx = pb[i * 3 + 0] - cx, dy = pb[i * 3 + 1] - cy, dz = pb[i * 3 + 2] - cz;
    d2s[i] = dx * dx + dy * dy + dz * dz;
  }
  if (tid == 0) { cnt1 = 0; cnt2 = 0; }
  __syncthreads();

  // V = bit pattern of the 64th smallest d2 (d2 >= 0 -> bits are order-preserving)
  unsigned lo = 0u, hi = 0x41000000u;   // 8.0f upper bound (max d2 = 3.0)
  for (int it = 0; it < 31; ++it) {
    unsigned mid = lo + ((hi - lo) >> 1);
    int cnt = 0;
#pragma unroll 4
    for (int k = 0; k < 32; ++k) {
      unsigned key = __float_as_uint(d2s[tid + k * 128]);
      cnt += (key <= mid) ? 1 : 0;
    }
#pragma unroll
    for (int off = 16; off > 0; off >>= 1) cnt += __shfl_xor(cnt, off, 32);
    if (lane == 0) redC[wave] = cnt;
    __syncthreads();
    int tot = redC[0] + redC[1] + redC[2] + redC[3];
    __syncthreads();
    if (tot >= KNBR) hi = mid; else lo = mid + 1;
  }
  const unsigned V = hi;

  int*   nI = nbrIdx + (size_t)c * KNBR;
  float* nD = nbrD2  + (size_t)c * KNBR;
  for (int k = 0; k < 32; ++k) {
    int i = tid + k * 128;
    if (__float_as_uint(d2s[i]) < V) {
      int p = atomicAdd(&cnt1, 1);
      nI[p] = i; nD[p] = d2s[i];
    }
  }
  __syncthreads();
  const int n1 = cnt1;
  for (int k = 0; k < 32; ++k) {
    int i = tid + k * 128;
    if (__float_as_uint(d2s[i]) == V) {
      int q = atomicAdd(&cnt2, 1);
      if (n1 + q < KNBR) { nI[n1 + q] = i; nD[n1 + q] = d2s[i]; }
    }
  }
}

// ================================================================ TDM helper
// 1-D tensor DMA: copy nElems f16 from global to LDS (wave-level op).
// This toolchain exposes the 6-arg builtin: (v4u, v8i, v4i, v4i, v8i, i32 cpol).
__device__ __forceinline__
void tdm_load_1d(const void* gptr, unsigned lds_off, unsigned nElems) {
  unsigned long long ga = (unsigned long long)(uintptr_t)gptr;
  v4u g0;
  g0[0] = 1u;                                       // count=1, user descriptor
  g0[1] = lds_off;                                  // lds_addr
  g0[2] = (unsigned)(ga & 0xFFFFFFFFu);             // global_addr lo
  g0[3] = (unsigned)((ga >> 32) & 0x01FFFFFFu) | (2u << 30);  // hi + type=2
  v8i g1;
  g1[0] = (int)(1u << 16);                          // data_size=1 (2 bytes)
  g1[1] = (int)((nElems & 0xFFFFu) << 16);          // tensor_dim0 lo16
  g1[2] = (int)(((nElems >> 16) & 0xFFFFu) | (1u << 16)); // dim0 hi | tensor_dim1=1
  g1[3] = (int)((nElems & 0xFFFFu) << 16);          // tile_dim0 = nElems (<=65535)
  g1[4] = 0;                                        // tile_dim1/2 unused
  g1[5] = (int)nElems;                              // tensor_dim0_stride lo
  g1[6] = 0;
  g1[7] = 0;
  v4i g2 = {0, 0, 0, 0};
  v4i g3 = {0, 0, 0, 0};
  v8i g4 = {0, 0, 0, 0, 0, 0, 0, 0};                // extra group (unused, zero)
  __builtin_amdgcn_tensor_load_to_lds(g0, g1, g2, g3, g4, 0);
}

// ================================================================ fused MSG-MLP
struct MlpArgs {
  const float*    x;
  const float*    pos;
  const float*    posOut;
  const int*      nbrIdx;
  const float*    nbrD2;
  const _Float16* wpack;
  const float*    bias[9];
  float*          out;
};

__global__ __launch_bounds__(128)
void mlp_kernel(MlpArgs a) {
  __shared__ alignas(16) _Float16 act0[64][136];
  __shared__ alignas(16) _Float16 act1[64][136];
  __shared__ alignas(16) _Float16 wbuf[96 * 128];   // swizzled B fragments
  __shared__ int valid[64];

  const int c  = blockIdx.x;
  const int br = blockIdx.y;
  const int b  = c >> 10;
  const int tid  = threadIdx.x;
  const int lane = tid & 31, wave = tid >> 5;

  // ---- build grouped features: [64 rows x 32 cols(f16)], cols 0..5 live
  const float cpx = a.posOut[c * 3 + 0];
  const float cpy = a.posOut[c * 3 + 1];
  const float cpz = a.posOut[c * 3 + 2];
  if (tid < 64) {
    int j = a.nbrIdx[(size_t)c * KNBR + tid];
    const float* pj = a.pos + (size_t)(b * NPER + j) * 3;
    const float* xj = a.x   + (size_t)(b * NPER + j) * 3;
    act0[tid][0] = (_Float16)xj[0];
    act0[tid][1] = (_Float16)xj[1];
    act0[tid][2] = (_Float16)xj[2];
    act0[tid][3] = (_Float16)(pj[0] - cpx);
    act0[tid][4] = (_Float16)(pj[1] - cpy);
    act0[tid][5] = (_Float16)(pj[2] - cpz);
#pragma unroll
    for (int q = 6; q < 32; ++q) act0[tid][q] = (_Float16)0.0f;
    valid[tid] = (a.nbrD2[(size_t)c * KNBR + tid] <= R2T[br]) ? 1 : 0;
  }

  _Float16 (*A)[136]  = act0;
  _Float16 (*Hn)[136] = act1;

  for (int l = 0; l < 3; ++l) {
    const int Kp  = KpadT[br][l];
    const int Cn  = CoutT[br][l];
    const int nks = Kp >> 5;
    __syncthreads();                       // prior wbuf reads / act writes done
    if (tid < 32) {                        // wave 0 drives the TDM
      unsigned lds_off = (unsigned)(uintptr_t)(void*)&wbuf[0];
      tdm_load_1d(a.wpack + WoffT[br][l], lds_off, (unsigned)(Kp * Cn));
      __builtin_amdgcn_s_wait_tensorcnt(0);
    }
    __syncthreads();                       // weights visible to all waves

    const float* bias = a.bias[br * 3 + l];
    const int nct = Cn >> 4;               // 16-wide col tiles
    for (int ct = wave; ct < nct; ct += 4) {
      v8f acc[4];
#pragma unroll
      for (int rt = 0; rt < 4; ++rt) acc[rt] = (v8f)0.0f;

      for (int ks = 0; ks < nks; ++ks) {
        const int k0 = ks * 32;
        // B fragment: lane-contiguous 32B in swizzled wbuf -> 2x ds_load_b128
        const int fbase = ((ct * nks + ks) << 9) + (lane << 4);
        v8h blo = *reinterpret_cast<const v8h*>(&wbuf[fbase]);
        v8h bhi = *reinterpret_cast<const v8h*>(&wbuf[fbase + 8]);
        v16h bfrag = __builtin_shufflevector(blo, bhi,
            0, 1, 2, 3, 4, 5, 6, 7, 8, 9, 10, 11, 12, 13, 14, 15);

#pragma unroll
        for (int rt = 0; rt < 4; ++rt) {
          // A fragment (16x32 f16): two 16B LDS loads per lane
          const int row = rt * 16 + (lane & 15);
          const int hof = (lane >> 4) * 16;
          v8h alo = *reinterpret_cast<const v8h*>(&A[row][k0 + hof]);
          v8h ahi = *reinterpret_cast<const v8h*>(&A[row][k0 + hof + 8]);
          v16h afrag = __builtin_shufflevector(alo, ahi,
              0, 1, 2, 3, 4, 5, 6, 7, 8, 9, 10, 11, 12, 13, 14, 15);
          acc[rt] = __builtin_amdgcn_wmma_f32_16x16x32_f16(
              false, afrag, false, bfrag, (short)0, acc[rt], false, false);
        }
      }
      // epilogue: +bias, ReLU, f16 into next activation buffer
      const int col   = ct * 16 + (lane & 15);
      const float bv  = bias[col];
      const int rbase = (lane >> 4) * 8;
#pragma unroll
      for (int rt = 0; rt < 4; ++rt) {
#pragma unroll
        for (int r = 0; r < 8; ++r) {
          float v = acc[rt][r] + bv;
          v = v > 0.0f ? v : 0.0f;
          Hn[rt * 16 + rbase + r][col] = (_Float16)v;
        }
      }
    }
    { _Float16 (*t)[136] = A; A = Hn; Hn = t; }
  }
  __syncthreads();

  // ---- masked max-pool over the 64 neighbor rows
  const int Cnf = CoutT[br][2];
  for (int col = tid; col < Cnf; col += 128) {
    float m = -__builtin_inff(); int any = 0;
    for (int k = 0; k < KNBR; ++k) {
      if (valid[k]) { float v = (float)A[k][col]; m = v > m ? v : m; any = 1; }
    }
    a.out[(size_t)c * COUT_TOTAL + CoffT[br] + col] = any ? m : 0.0f;
  }
}

// ================================================================ launch
extern "C" void kernel_launch(void* const* d_in, const int* in_sizes, int n_in,
                              void* d_out, int out_size, void* d_ws, size_t ws_size,
                              hipStream_t stream) {
  const float* x   = (const float*)d_in[0];
  const float* pos = (const float*)d_in[1];

  float* out      = (float*)d_out;
  float* posOut   = out + (size_t)NCENT * COUT_TOTAL;
  float* batchOut = posOut + (size_t)NCENT * 3;

  // workspace layout
  char* w = (char*)d_ws;
  int*   nbrIdx = (int*)w;    w += (size_t)NCENT * KNBR * sizeof(int);
  float* nbrD2  = (float*)w;  w += (size_t)NCENT * KNBR * sizeof(float);
  _Float16* wpack = (_Float16*)w;   // WPACK_HALVES halves

  // 1) farthest point sampling (also emits pos_out / batch_out)
  fps_kernel<<<Bc, 256, 0, stream>>>(pos, posOut, batchOut);

  // 2) weight conversion + fragment swizzle (independent of fps)
  W9 w9;
  for (int br = 0; br < 3; ++br)
    for (int l = 0; l < 3; ++l)
      w9.w[br * 3 + l] = (const float*)d_in[3 + br * 6 + l * 2];
  prep_weights<<<9, 256, 0, stream>>>(w9, wpack);

  // 3) 64-NN per centroid (serves all three radii via d2 thresholds)
  knn_kernel<<<NCENT, 128, 0, stream>>>(pos, posOut, nbrIdx, nbrD2);

  // 4) fused grouped-MLP + masked max-pool, one branch per blockIdx.y
  MlpArgs ma;
  ma.x = x; ma.pos = pos; ma.posOut = posOut;
  ma.nbrIdx = nbrIdx; ma.nbrD2 = nbrD2; ma.wpack = wpack; ma.out = out;
  for (int br = 0; br < 3; ++br)
    for (int l = 0; l < 3; ++l)
      ma.bias[br * 3 + l] = (const float*)d_in[3 + br * 6 + l * 2 + 1];
  mlp_kernel<<<dim3(NCENT, 3), 128, 0, stream>>>(ma);
}
